// YawNetVLAD_13280038880042
// MI455X (gfx1250) — compile-verified, matrix operand order checked by reading, not run on hardware
//
#include <hip/hip_runtime.h>
#include <hip/hip_bf16.h>

// ---------------------------------------------------------------------------
// YawNetVLAD on MI455X (gfx1250): f16 WMMA for all GEMM-shaped work.
//   N=1, C=64, H=48, W=900, K=64, OUT=256
// ---------------------------------------------------------------------------

typedef __attribute__((ext_vector_type(16))) _Float16 v16h;
typedef __attribute__((ext_vector_type(8)))  float    v8f;

union FragU { v16h v; _Float16 h[16]; unsigned int u[8]; };

#define CH   64
#define HH   48
#define WW   900
#define KK   64
#define HWP  43200         // 48*900

// K-index of the 2-element pair held in VGPR j for lane-half `off` (0 or 8).
// 16-bit A layout (16x32): VGPR0..3 -> K = off+2j ; VGPR4..7 -> K = 16+off+2(j-4).
__device__ __forceinline__ int kpos(int j, int off) {
    return (j < 4) ? (2 * j + off) : (16 + 2 * (j - 4) + off);
}

__device__ __forceinline__ v8f wmma_f16(const FragU& A, const FragU& B, v8f c) {
    return __builtin_amdgcn_wmma_f32_16x16x32_f16(false, A.v, false, B.v,
                                                  (short)0, c, false, false);
}

// ---------------------------------------------------------------------------
// Kernel 1: per-w-column block (900 blocks x 128 threads = 4 waves).
//   GEMM1: s[h,k] = x[h,:]·conv_w[k,:] + b ; xc[h,k] = x[h,:]·cent[k,:]
//   softmax over k -> a ; pmap[k,h,w] = sqrt(relu(|x|^2+|c|^2-2xc))
//   GEMM2: res[k,c] = sum_h a[h,k] x[h,c] - cent[k,c]*sum_h a[h,k]
// ---------------------------------------------------------------------------
__global__ void k_assign(const float* __restrict__ x, const float* __restrict__ cent,
                         const float* __restrict__ convw, const float* __restrict__ convb,
                         _Float16* __restrict__ pmap, float* __restrict__ res)
{
    __shared__ _Float16 xcol[64][64];   // [h pad to 64][c]
    __shared__ _Float16 a_lds[64][64];  // [h pad to 64][k]
    __shared__ _Float16 cw_lds[64][64]; // [k][c]
    __shared__ _Float16 ct_lds[64][64]; // [k][c]
    __shared__ float x2s[64], c2s[64], ahsum[64];

    const int w    = blockIdx.x;
    const int tid  = threadIdx.x;
    const int lane = tid & 31;
    const int wav  = tid >> 5;

    // ---- stage operands (f32 global -> f16 LDS) ----
    for (int i = tid; i < 4096; i += 128) {
        int h = i >> 6, c = i & 63;
        float f = 0.f;
        if (h < HH) f = x[(c * HH + h) * WW + w];
        xcol[h][c]  = (_Float16)f;
        a_lds[h][c] = (_Float16)0.f;
        cw_lds[h][c] = (_Float16)convw[i]; // reuse i as k*64+c
        ct_lds[h][c] = (_Float16)cent[i];
    }
    if (tid < 64) {                       // c2[k]
        float s = 0.f;
        for (int c = 0; c < 64; ++c) { float v = cent[tid * 64 + c]; s += v * v; }
        c2s[tid] = s;
    } else if (tid < 128) {               // x2[h]
        int h = tid - 64; float s = 0.f;
        if (h < HH) for (int c = 0; c < 64; ++c) {
            float v = x[(c * HH + h) * WW + w]; s += v * v;
        }
        x2s[h] = s;
    }
    __syncthreads();

    // ---- GEMM1: wave `wav` owns h-rows [16*wav, 16*wav+16) ----
    const int off  = (lane < 16) ? 0 : 8;
    const int mrow = (lane & 15) + 16 * wav;          // A row (h)
    v8f accS[4] = {}, accX[4] = {};
    for (int ks = 0; ks < 2; ++ks) {
        FragU A;
#pragma unroll
        for (int j = 0; j < 8; ++j)
            A.u[j] = *(const unsigned*)&xcol[mrow][kpos(j, off) + 32 * ks];
#pragma unroll
        for (int t = 0; t < 4; ++t) {
            int n = (lane & 15) + 16 * t;
            FragU Bs, Bx;
#pragma unroll
            for (int j = 0; j < 8; ++j) {
                int c0 = kpos(j, off) + 32 * ks;
                Bs.u[j] = *(const unsigned*)&cw_lds[n][c0];
                Bx.u[j] = *(const unsigned*)&ct_lds[n][c0];
            }
            accS[t] = wmma_f16(A, Bs, accS[t]);
            accX[t] = wmma_f16(A, Bx, accX[t]);
        }
    }
#pragma unroll
    for (int t = 0; t < 4; ++t) {
        float cb = convb[(lane & 15) + 16 * t];
#pragma unroll
        for (int g = 0; g < 8; ++g) accS[t][g] += cb;
    }

    // ---- softmax over k (cross-lane within 16-lane half + 4 N-tiles) ----
#pragma unroll
    for (int g = 0; g < 8; ++g) {
        float mv = accS[0][g];
#pragma unroll
        for (int t = 1; t < 4; ++t) mv = fmaxf(mv, accS[t][g]);
        for (int d = 1; d < 16; d <<= 1) mv = fmaxf(mv, __shfl_xor(mv, d, 32));
        float sv = 0.f;
#pragma unroll
        for (int t = 0; t < 4; ++t) {
            float e = __expf(accS[t][g] - mv);
            accS[t][g] = e; sv += e;
        }
        for (int d = 1; d < 16; d <<= 1) sv += __shfl_xor(sv, d, 32);
        float inv = 1.f / sv;
#pragma unroll
        for (int t = 0; t < 4; ++t) accS[t][g] *= inv;
    }

    // ---- store a -> LDS ; pmap -> global ----
#pragma unroll
    for (int t = 0; t < 4; ++t) {
        int n = (lane & 15) + 16 * t;                 // cluster k
#pragma unroll
        for (int g = 0; g < 8; ++g) {
            int m = g + ((lane >= 16) ? 8 : 0) + 16 * wav;  // h
            if (m < HH) {
                a_lds[m][n] = (_Float16)accS[t][g];
                float d2 = x2s[m] + c2s[n] - 2.f * accX[t][g];
                pmap[(n * HH + m) * WW + w] = (_Float16)sqrtf(fmaxf(d2, 0.f));
            }
        }
    }
    __syncthreads();
    if (tid < 64) {                                   // deterministic a_hsum[k]
        float s = 0.f;
        for (int h = 0; h < HH; ++h) s += (float)a_lds[h][tid];
        ahsum[tid] = s;
    }
    __syncthreads();

    // ---- GEMM2: res[k,c] ; wave owns k-rows [16*wav,16*wav+16) ----
    v8f accR[4] = {};
    const int mk = (lane & 15) + 16 * wav;            // A row (k)
    for (int ks = 0; ks < 2; ++ks) {
        FragU A;
#pragma unroll
        for (int j = 0; j < 8; ++j) {
            int h0 = kpos(j, off) + 32 * ks;
            A.h[2 * j]     = a_lds[h0][mk];
            A.h[2 * j + 1] = a_lds[h0 + 1][mk];
        }
#pragma unroll
        for (int t = 0; t < 4; ++t) {
            int n = (lane & 15) + 16 * t;             // channel c
            FragU B;
#pragma unroll
            for (int j = 0; j < 8; ++j) {
                int h0 = kpos(j, off) + 32 * ks;
                B.h[2 * j]     = xcol[h0][n];
                B.h[2 * j + 1] = xcol[h0 + 1][n];
            }
            accR[t] = wmma_f16(A, B, accR[t]);
        }
    }
#pragma unroll
    for (int t = 0; t < 4; ++t) {
        int cc = (lane & 15) + 16 * t;
#pragma unroll
        for (int g = 0; g < 8; ++g) {
            int kk = g + ((lane >= 16) ? 8 : 0) + 16 * wav;
            float v = accR[t][g] - cent[kk * 64 + cc] * ahsum[kk];
            res[(kk * 64 + cc) * WW + w] = v;
        }
    }
}

// ---------------------------------------------------------------------------
// Kernel 2: circular sliding-window sums -> vlad[k*64+c][90]
// ---------------------------------------------------------------------------
__global__ void k_vlad(const float* __restrict__ res, float* __restrict__ vlad)
{
    int idx = blockIdx.x * blockDim.x + threadIdx.x;
    if (idx >= 64 * 64 * 90) return;
    int kc = idx / 90, y = idx % 90;
    const float* row = res + kc * WW;
    int start = y * 10 - 100;
    float s = 0.f;
    for (int j = 0; j < 200; ++j) s += row[(start + j + 900) % 900];
    vlad[kc * 90 + y] = s;
}

// ---------------------------------------------------------------------------
// Kernel 3: convert conv weights f32 [co][ci][5][5] -> f16 [tap][co][ci]
// ---------------------------------------------------------------------------
__global__ void k_cvtw(const float* __restrict__ in, _Float16* __restrict__ out)
{
    int e = blockIdx.x * blockDim.x + threadIdx.x;
    if (e >= 64 * 64 * 25) return;
    int co = e / 1600, r = e % 1600, ci = r / 25, tap = r % 25;
    out[(tap * 64 + co) * 64 + ci] = (_Float16)in[e];
}

// ---------------------------------------------------------------------------
// Kernel 4: conv1 (1 in-ch, direct) + BN + LeakyReLU, per image
// ---------------------------------------------------------------------------
__global__ void k_conv1(const _Float16* __restrict__ in, const float* __restrict__ wgt,
                        const float* __restrict__ bias,
                        const float* __restrict__ g, const float* __restrict__ bb,
                        const float* __restrict__ m, const float* __restrict__ v,
                        _Float16* __restrict__ out)
{
    int o = blockIdx.x * blockDim.x + threadIdx.x;
    if (o >= 64 * HWP) return;
    int co = o / HWP, r = o % HWP, h = r / WW, wx = r % WW;
    float acc = bias[co];
    for (int dy = 0; dy < 5; ++dy) {
        int gh = h + dy - 2; if (gh < 0 || gh >= HH) continue;
        for (int dx = 0; dx < 5; ++dx) {
            int gw = wx + dx - 2; if (gw < 0 || gw >= WW) continue;
            acc += wgt[co * 25 + dy * 5 + dx] * (float)in[gh * WW + gw];
        }
    }
    float s = g[co] * rsqrtf(v[co] + 1e-5f);
    float val = acc * s + (bb[co] - m[co] * s);
    val = (val >= 0.f) ? val : 0.2f * val;
    out[(co * HH + h) * WW + wx] = (_Float16)val;
}

// ---------------------------------------------------------------------------
// Kernel 5: 5x5 conv, 64->64 channels, implicit GEMM on WMMA f16.
//   grid(29,6), 256 threads (8 waves). Block tile 8h x 32w = 256 pixels.
//   Each wave owns one h-row (2 M-tiles of 16 px) so every B (weight)
//   fragment feeds TWO wmma ops. Inner dim = 64ci x 25 taps.
// ---------------------------------------------------------------------------
__global__ void k_conv5(const _Float16* __restrict__ in, const _Float16* __restrict__ wgt16,
                        const float* __restrict__ bias,
                        const float* __restrict__ g, const float* __restrict__ bb,
                        const float* __restrict__ m, const float* __restrict__ v,
                        int use_bn, int use_lrelu, _Float16* __restrict__ out)
{
    __shared__ _Float16 in_lds[12][36][64];  // [h+halo][w+halo][ci] (ci contiguous)
    __shared__ float bns[64], bnt[64];

    const int tid = threadIdx.x;
    for (int co = tid; co < 64; co += 256) {
        float s = 1.f, t = 0.f;
        if (use_bn) { s = g[co] * rsqrtf(v[co] + 1e-5f); t = bb[co] - m[co] * s; }
        bns[co] = s; bnt[co] = t;
    }
    // warm L2 with the weight block (gfx1250 global_prefetch)
    for (int cl = tid; cl < 1600; cl += 256)
        __builtin_prefetch(wgt16 + cl * 64, 0, 0);

    const int h0 = blockIdx.y * 8 - 2;
    const int w0 = blockIdx.x * 32 - 2;
    for (int e = tid; e < 64 * 12 * 36; e += 256) {
        int ci = e / 432, r = e % 432, hh = r / 36, ww = r % 36;
        int gh = h0 + hh, gw = w0 + ww;
        _Float16 val = (_Float16)0.f;
        if (gh >= 0 && gh < HH && gw >= 0 && gw < WW)
            val = in[(ci * HH + gh) * WW + gw];
        in_lds[hh][ww][ci] = val;
    }
    __syncthreads();

    const int wav  = tid >> 5;                    // wave owns image row h = wav
    const int lane = tid & 31;
    const int off  = (lane < 16) ? 0 : 8;
    const int wl0  = lane & 15;                   // M-tile 0: w in [0,16)
    const int wl1  = 16 + (lane & 15);            // M-tile 1: w in [16,32)

    v8f acc[2][4] = {};
    for (int tap = 0; tap < 25; ++tap) {
        int dy = tap / 5, dx = tap % 5;
        const _Float16* arow0 = &in_lds[wav + dy][wl0 + dx][0];
        const _Float16* arow1 = &in_lds[wav + dy][wl1 + dx][0];
        const _Float16* wrow  = wgt16 + tap * 4096;
        for (int ks = 0; ks < 2; ++ks) {
            FragU A0, A1;
#pragma unroll
            for (int j = 0; j < 8; ++j) {
                int c0 = kpos(j, off) + 32 * ks;
                A0.u[j] = *(const unsigned*)(arow0 + c0);
                A1.u[j] = *(const unsigned*)(arow1 + c0);
            }
#pragma unroll
            for (int t = 0; t < 4; ++t) {
                int n = (lane & 15) + 16 * t;     // out channel
                FragU B;
#pragma unroll
                for (int j = 0; j < 8; ++j)
                    B.u[j] = *(const unsigned*)(wrow + n * 64 + kpos(j, off) + 32 * ks);
                acc[0][t] = wmma_f16(A0, B, acc[0][t]);
                acc[1][t] = wmma_f16(A1, B, acc[1][t]);
            }
        }
    }
#pragma unroll
    for (int t = 0; t < 4; ++t) {
        int co = (lane & 15) + 16 * t;
        float s = bns[co], tt = bnt[co], cb = bias[co];
#pragma unroll
        for (int i = 0; i < 2; ++i) {
#pragma unroll
            for (int gg = 0; gg < 8; ++gg) {
                int pw = i * 16 + gg + ((lane >= 16) ? 8 : 0);   // w within tile
                int gh = blockIdx.y * 8 + wav;
                int gw = blockIdx.x * 32 + pw;
                if (gw < WW) {
                    float val = (acc[i][t][gg] + cb) * s + tt;
                    if (use_lrelu) val = (val >= 0.f) ? val : 0.2f * val;
                    out[(co * HH + gh) * WW + gw] = (_Float16)val;
                }
            }
        }
    }
}

// ---------------------------------------------------------------------------
// Kernel 6: column max over h ; Kernel 7: circular window max over w
// ---------------------------------------------------------------------------
__global__ void k_colmax(const _Float16* __restrict__ f, float* __restrict__ colmax)
{
    int idx = blockIdx.x * blockDim.x + threadIdx.x;
    if (idx >= 64 * WW) return;
    int c = idx / WW, wx = idx % WW;
    float mv = -3.4e38f;
    for (int h = 0; h < HH; ++h) mv = fmaxf(mv, (float)f[(c * HH + h) * WW + wx]);
    colmax[idx] = mv;
}

__global__ void k_mapwin(const float* __restrict__ colmax, float* __restrict__ mapf, int b)
{
    int idx = blockIdx.x * blockDim.x + threadIdx.x;
    if (idx >= 64 * 90) return;
    int c = idx / 90, y = idx % 90;
    int start = y * 10 - 100;
    float mv = -3.4e38f;
    for (int j = 0; j < 200; ++j)
        mv = fmaxf(mv, colmax[c * WW + (start + j + 900) % 900]);
    mapf[(b * 64 + c) * 90 + y] = mv;
}

// ---------------------------------------------------------------------------
// Kernel 8: per-yaw intra-norm + L2 norm + MLP + final L2 norm
// ---------------------------------------------------------------------------
__global__ void k_head(const float* __restrict__ vlad, const float* __restrict__ mapf,
                       const float* __restrict__ mlpw, const float* __restrict__ mlpb,
                       float* __restrict__ outp)
{
    __shared__ float feat[8192];
    __shared__ float red[256];
    __shared__ float outv[256];
    const int y = blockIdx.x, tid = threadIdx.x;

    for (int i = tid; i < 4096; i += 256) {
        feat[i]        = vlad[i * 90 + y];
        feat[4096 + i] = mapf[i * 90 + y];
    }
    __syncthreads();
    for (int gk = tid; gk < 128; gk += 256) {        // intra-norm per cluster
        int base = gk * 64; float s = 0.f;
        for (int c = 0; c < 64; ++c) s += feat[base + c] * feat[base + c];
        float inv = 1.f / fmaxf(sqrtf(s), 1e-12f);
        for (int c = 0; c < 64; ++c) feat[base + c] *= inv;
    }
    __syncthreads();
    float s1 = 0.f, s2 = 0.f;
    for (int i = tid; i < 4096; i += 256) {
        s1 += feat[i] * feat[i];
        s2 += feat[4096 + i] * feat[4096 + i];
    }
    red[tid] = s1; __syncthreads();
    for (int st = 128; st > 0; st >>= 1) { if (tid < st) red[tid] += red[tid + st]; __syncthreads(); }
    float n1 = red[0]; __syncthreads();
    red[tid] = s2; __syncthreads();
    for (int st = 128; st > 0; st >>= 1) { if (tid < st) red[tid] += red[tid + st]; __syncthreads(); }
    float n2 = red[0]; __syncthreads();
    float i1 = 1.f / fmaxf(sqrtf(n1), 1e-12f);
    float i2 = 1.f / fmaxf(sqrtf(n2), 1e-12f);
    for (int i = tid; i < 4096; i += 256) { feat[i] *= i1; feat[4096 + i] *= i2; }
    __syncthreads();

    {   // MLP: one output per thread
        float acc = mlpb[tid];
        const float* wr = mlpw + tid * 8192;
        for (int i = 0; i < 8192; ++i) acc += feat[i] * wr[i];
        outv[tid] = acc;
        red[tid]  = acc * acc;
    }
    __syncthreads();
    for (int st = 128; st > 0; st >>= 1) { if (tid < st) red[tid] += red[tid + st]; __syncthreads(); }
    float inv = 1.f / fmaxf(sqrtf(red[0]), 1e-12f);
    outp[y * 256 + tid] = outv[tid] * inv;
}

// ---------------------------------------------------------------------------
// Workspace layout (bytes, 512-aligned); total ~40.5 MB
// ---------------------------------------------------------------------------
#define OFF_PMAP   0u                    // f16 64*48*900      = 5,529,600 B
#define OFF_RES    5529600u              // f32 64*64*900      = 14,745,600 B
#define OFF_VLAD   20275200u             // f32 64*64*90       = 1,474,560 B
#define OFF_WG2    21749760u             // f16 25*64*64       = 204,800 B
#define OFF_WG3    21954560u             // f16 25*64*64       = 204,800 B
#define OFF_T1     22159360u             // f16 64*48*900      = 5,529,600 B
#define OFF_T2     27688960u
#define OFF_T3     33218560u
#define OFF_CMAX   38748160u             // f32 64*900         = 230,400 B
#define OFF_MAP    38978560u             // f32 64*64*90       = 1,474,560 B

extern "C" void kernel_launch(void* const* d_in, const int* in_sizes, int n_in,
                              void* d_out, int out_size, void* d_ws, size_t ws_size,
                              hipStream_t stream)
{
    (void)in_sizes; (void)n_in; (void)out_size; (void)ws_size;
    const float* x     = (const float*)d_in[0];
    const float* cent  = (const float*)d_in[1];
    const float* convw = (const float*)d_in[2];
    const float* convb = (const float*)d_in[3];
    const float* mc1w  = (const float*)d_in[4];
    const float* mc1b  = (const float*)d_in[5];
    const float* bn1g  = (const float*)d_in[6];
    const float* bn1b  = (const float*)d_in[7];
    const float* bn1m  = (const float*)d_in[8];
    const float* bn1v  = (const float*)d_in[9];
    const float* mc2w  = (const float*)d_in[10];
    const float* mc2b  = (const float*)d_in[11];
    const float* bn2g  = (const float*)d_in[12];
    const float* bn2b  = (const float*)d_in[13];
    const float* bn2m  = (const float*)d_in[14];
    const float* bn2v  = (const float*)d_in[15];
    const float* mc3w  = (const float*)d_in[16];
    const float* mc3b  = (const float*)d_in[17];
    const float* mlpw  = (const float*)d_in[18];
    const float* mlpb  = (const float*)d_in[19];

    char* ws = (char*)d_ws;
    _Float16* pmap  = (_Float16*)(ws + OFF_PMAP);
    float*    res   = (float*)   (ws + OFF_RES);
    float*    vlad  = (float*)   (ws + OFF_VLAD);
    _Float16* wg2   = (_Float16*)(ws + OFF_WG2);
    _Float16* wg3   = (_Float16*)(ws + OFF_WG3);
    _Float16* t1    = (_Float16*)(ws + OFF_T1);
    _Float16* t2    = (_Float16*)(ws + OFF_T2);
    _Float16* t3    = (_Float16*)(ws + OFF_T3);
    float*    cmax  = (float*)   (ws + OFF_CMAX);
    float*    mapf  = (float*)   (ws + OFF_MAP);

    // assign + pmap + residual (WMMA), then VLAD windows
    k_assign<<<WW, 128, 0, stream>>>(x, cent, convw, convb, pmap, res);
    k_vlad<<<(64 * 64 * 90 + 255) / 256, 256, 0, stream>>>(res, vlad);

    // weight conversion for the two heavy convs
    k_cvtw<<<(64 * 64 * 25 + 255) / 256, 256, 0, stream>>>(mc2w, wg2);
    k_cvtw<<<(64 * 64 * 25 + 255) / 256, 256, 0, stream>>>(mc3w, wg3);

    dim3 cgrid(29, 6);
    for (int b = 0; b < 64; ++b) {
        const _Float16* img = pmap + (size_t)b * HWP;
        k_conv1<<<(64 * HWP + 255) / 256, 256, 0, stream>>>(
            img, mc1w, mc1b, bn1g, bn1b, bn1m, bn1v, t1);
        k_conv5<<<cgrid, 256, 0, stream>>>(
            t1, wg2, mc2b, bn2g, bn2b, bn2m, bn2v, 1, 1, t2);
        k_conv5<<<cgrid, 256, 0, stream>>>(
            t2, wg3, mc3b, nullptr, nullptr, nullptr, nullptr, 0, 0, t3);
        k_colmax<<<(64 * WW + 255) / 256, 256, 0, stream>>>(t3, cmax);
        k_mapwin<<<(64 * 90 + 255) / 256, 256, 0, stream>>>(cmax, mapf, b);
    }

    k_head<<<90, 256, 0, stream>>>(vlad, mapf, mlpw, mlpb, (float*)d_out);
}
